// GlobalMixer_86311662780603
// MI455X (gfx1250) — matrix operand order repaired and checked by reading, not compiled, not run on hardware
//
#include <hip/hip_runtime.h>
#include <hip/hip_bf16.h>
#include <math.h>

typedef __attribute__((ext_vector_type(16))) _Float16 v16h;
typedef __attribute__((ext_vector_type(8)))  _Float16 v8h;
typedef __attribute__((ext_vector_type(8)))  float    v8f;

#define BATCH 4
#define SEQ   2048
#define DIM   768
#define NHEAD 12
#define HDIM  64
#define SCALE 0.125f
#define MTOT  (BATCH*SEQ)   /* 8192 */
#define QKVN  (3*DIM)       /* 2304 */

static __device__ __forceinline__ v8f wmma_f16(v16h a, v16h b, v8f c) {
  // D = A(16x32 f16) * B(32x16 f16) + C(16x16 f32)
  return __builtin_amdgcn_wmma_f32_16x16x32_f16(false, a, false, b, (short)0, c,
                                                false, false);
}

// A-tile (16x32 f16) from row-major f16: row = l16; elements 0..7 <- K=hi*8..,
// elements 8..15 <- K=16+hi*8.. (CDNA5 16-bit A layout). 16B-aligned chunks.
static __device__ __forceinline__ v16h load_a16(const _Float16* row, int hi) {
  v16h a;
  *(v8h*)&a       = *(const v8h*)(row + hi * 8);
  *((v8h*)&a + 1) = *(const v8h*)(row + 16 + hi * 8);
  return a;
}

// B-tile (32x16 f16) from row-major f16 [K][N] (row stride ld): p -> column
// N = l16 at K=0; elements 0..7 <- K=hi*8.., elements 8..15 <- K=16+hi*8..
static __device__ __forceinline__ v16h load_b16(const _Float16* p, int hi, size_t ld) {
  v16h b;
#pragma unroll
  for (int s = 0; s < 8; ++s) {
    b[s]     = p[(size_t)(hi * 8 + s) * ld];
    b[8 + s] = p[(size_t)(16 + hi * 8 + s) * ld];
  }
  return b;
}

// ---------------------------------------------------------------------------
// Kernel 0: one-shot f32 -> f16 conversion (8 elements / thread, b128 I/O).
// ---------------------------------------------------------------------------
__global__ void __launch_bounds__(256)
cvt_f16(const float* __restrict__ in, _Float16* __restrict__ out) {
  const size_t i = ((size_t)blockIdx.x * 256 + threadIdx.x) * 8;
  const v8f x = *(const v8f*)(in + i);
  v8h h;
#pragma unroll
  for (int j = 0; j < 8; ++j) h[j] = (_Float16)x[j];
  *(v8h*)(out + i) = h;
}

// ---------------------------------------------------------------------------
// Kernel 1: QKV projection, all-f16 operands. One wave -> 16x64 output tile
// (one A load feeds 4 WMMAs per K-step; 4 waves/block share B rows).
// Scatters to q [B,H,N,64], kT [B,H,64,N] (transposed), v [B,H,N,64].
// grid = (QKVN/64, MTOT/16/4), block = 128.
// ---------------------------------------------------------------------------
__global__ void __launch_bounds__(128)
qkv_gemm(const _Float16* __restrict__ X, const _Float16* __restrict__ W,
         _Float16* __restrict__ q, _Float16* __restrict__ kT,
         _Float16* __restrict__ v) {
  const int lane = threadIdx.x & 31;
  const int l16  = lane & 15;
  const int hi   = lane >> 4;
  const int mt   = blockIdx.y * 4 + (threadIdx.x >> 5);
  const int nt64 = blockIdx.x;                         // 0..35

  const _Float16* arow = X + (size_t)(mt * 16 + l16) * DIM;
  const _Float16* brow = W + nt64 * 64 + l16;

  v8f c0 = {}, c1 = {}, c2 = {}, c3 = {};
  for (int k0 = 0; k0 < DIM; k0 += 32) {
    const _Float16* bp = brow + (size_t)k0 * QKVN;
    __builtin_prefetch(bp + (size_t)32 * QKVN, 0, 1);
    const v16h a = load_a16(arow + k0, hi);
    c0 = wmma_f16(a, load_b16(bp,      hi, QKVN), c0);
    c1 = wmma_f16(a, load_b16(bp + 16, hi, QKVN), c1);
    c2 = wmma_f16(a, load_b16(bp + 32, hi, QKVN), c2);
    c3 = wmma_f16(a, load_b16(bp + 48, hi, QKVN), c3);
  }

  // 64-wide tile aligned to HDIM: single (s, h) slice. Wave-uniform -> SALU.
  const int s = nt64 / (DIM / HDIM);   // 0=q 1=k 2=v
  const int h = nt64 % (DIM / HDIM);
#pragma unroll
  for (int t = 0; t < 4; ++t) {
    const v8f c = (t == 0) ? c0 : (t == 1) ? c1 : (t == 2) ? c2 : c3;
    const int d = 16 * t + l16;
#pragma unroll
    for (int r = 0; r < 8; ++r) {
      const int row = mt * 16 + r + 8 * hi;
      const int b_  = row >> 11;              // / SEQ
      const int tok = row & (SEQ - 1);
      const size_t bh = (size_t)(b_ * NHEAD + h);
      const _Float16 val = (_Float16)c[r];
      if (s == 0)      q [(bh * SEQ + tok) * HDIM + d] = val;
      else if (s == 1) kT[(bh * HDIM + d) * SEQ + tok] = val;
      else             v [(bh * SEQ + tok) * HDIM + d] = val;
    }
  }
}

// ---------------------------------------------------------------------------
// Kernel 2: flash attention. One wave per (bh, 16-query tile); online softmax
// over 64 key-blocks of 32. grid = (SEQ/16/4, B*H), block = 128.
// ---------------------------------------------------------------------------
__global__ void __launch_bounds__(128)
attention(const _Float16* __restrict__ q, const _Float16* __restrict__ kT,
          const _Float16* __restrict__ v, _Float16* __restrict__ ao) {
  __shared__ __align__(32) _Float16 pstage[4][16 * 32];

  const int lane = threadIdx.x & 31;
  const int l16  = lane & 15;
  const int hi   = lane >> 4;
  const int warp = threadIdx.x >> 5;
  const int mt   = blockIdx.x * 4 + warp;
  const int bh   = blockIdx.y;

  const _Float16* qb = q  + ((size_t)bh * SEQ + mt * 16) * HDIM;
  const _Float16* kb = kT + (size_t)bh * HDIM * SEQ;
  const _Float16* vb = v  + (size_t)bh * SEQ * HDIM;

  const v16h aq0 = load_a16(qb + (size_t)l16 * HDIM, hi);       // d 0..31
  const v16h aq1 = load_a16(qb + (size_t)l16 * HDIM + 32, hi);  // d 32..63

  float mrow[8], lrow[8];
#pragma unroll
  for (int r = 0; r < 8; ++r) { mrow[r] = -3.0e38f; lrow[r] = 0.0f; }
  v8f o0 = {}, o1 = {}, o2 = {}, o3 = {};
  _Float16* pl = pstage[warp];

  for (int key0 = 0; key0 < SEQ; key0 += 32) {
    v8f s0 = {}, s1 = {};
    {
      const _Float16* kr = kb + key0 + l16;                     // d = 0, key = l16
      s0 = wmma_f16(aq0, load_b16(kr, hi, SEQ), s0);
      s1 = wmma_f16(aq0, load_b16(kr + 16, hi, SEQ), s1);
      kr += (size_t)32 * SEQ;                                   // d = 32
      s0 = wmma_f16(aq1, load_b16(kr, hi, SEQ), s0);
      s1 = wmma_f16(aq1, load_b16(kr + 16, hi, SEQ), s1);
    }
    // Row stats replicated per 16-lane half (C-layout row M = r + 8*hi).
#pragma unroll
    for (int r = 0; r < 8; ++r) {
      const float a0 = s0[r] * SCALE;
      const float a1 = s1[r] * SCALE;
      float mx = fmaxf(a0, a1);
#pragma unroll
      for (int off = 1; off < 16; off <<= 1)
        mx = fmaxf(mx, __shfl_xor(mx, off, 32));
      const float mnew = fmaxf(mrow[r], mx);
      const float corr = __expf(mrow[r] - mnew);
      const float p0 = __expf(a0 - mnew);
      const float p1 = __expf(a1 - mnew);
      float rs = p0 + p1;
#pragma unroll
      for (int off = 1; off < 16; off <<= 1)
        rs += __shfl_xor(rs, off, 32);
      lrow[r] = lrow[r] * corr + rs;
      mrow[r] = mnew;
      o0[r] = o0[r] * corr; o1[r] = o1[r] * corr;
      o2[r] = o2[r] * corr; o3[r] = o3[r] * corr;
      const int rowl = (r + 8 * hi) * 32 + l16;
      pl[rowl]      = (_Float16)p0;
      pl[rowl + 16] = (_Float16)p1;
    }
    __syncthreads();
    const v16h pa = load_a16(pl + l16 * 32, hi);   // P as 16x32 A-matrix
    const _Float16* vr = vb + (size_t)key0 * HDIM + l16;
    o0 = wmma_f16(pa, load_b16(vr,      hi, HDIM), o0);
    o1 = wmma_f16(pa, load_b16(vr + 16, hi, HDIM), o1);
    o2 = wmma_f16(pa, load_b16(vr + 32, hi, HDIM), o2);
    o3 = wmma_f16(pa, load_b16(vr + 48, hi, HDIM), o3);
    __syncthreads();
  }

  const int b_ = bh / NHEAD, h = bh % NHEAD;       // wave-uniform -> SALU
#pragma unroll
  for (int r = 0; r < 8; ++r) {
    const float inv = 1.0f / lrow[r];
    const size_t row = (size_t)b_ * SEQ + mt * 16 + r + 8 * hi;
    _Float16* op = ao + row * DIM + h * HDIM + l16;
    op[0]  = (_Float16)(o0[r] * inv);
    op[16] = (_Float16)(o1[r] * inv);
    op[32] = (_Float16)(o2[r] * inv);
    op[48] = (_Float16)(o3[r] * inv);
  }
}

// ---------------------------------------------------------------------------
// Kernel 3: output projection + bias, all-f16 operands, f32 result.
// One wave -> 16x64 tile. grid = (DIM/64, MTOT/16/4), block = 128.
// ---------------------------------------------------------------------------
__global__ void __launch_bounds__(128)
proj_gemm(const _Float16* __restrict__ A, const _Float16* __restrict__ W,
          const float* __restrict__ bias, float* __restrict__ out) {
  const int lane = threadIdx.x & 31;
  const int l16  = lane & 15;
  const int hi   = lane >> 4;
  const int mt   = blockIdx.y * 4 + (threadIdx.x >> 5);
  const int nt64 = blockIdx.x;                          // 0..11

  const _Float16* arow = A + (size_t)(mt * 16 + l16) * DIM;
  const _Float16* brow = W + nt64 * 64 + l16;

  v8f c0 = {}, c1 = {}, c2 = {}, c3 = {};
  for (int k0 = 0; k0 < DIM; k0 += 32) {
    const _Float16* bp = brow + (size_t)k0 * DIM;
    __builtin_prefetch(bp + (size_t)32 * DIM, 0, 1);
    const v16h a = load_a16(arow + k0, hi);
    c0 = wmma_f16(a, load_b16(bp,      hi, DIM), c0);
    c1 = wmma_f16(a, load_b16(bp + 16, hi, DIM), c1);
    c2 = wmma_f16(a, load_b16(bp + 32, hi, DIM), c2);
    c3 = wmma_f16(a, load_b16(bp + 48, hi, DIM), c3);
  }
#pragma unroll
  for (int t = 0; t < 4; ++t) {
    const v8f c = (t == 0) ? c0 : (t == 1) ? c1 : (t == 2) ? c2 : c3;
    const int col = nt64 * 64 + 16 * t + l16;
    const float bv = bias[col];
#pragma unroll
    for (int r = 0; r < 8; ++r) {
      const int row = mt * 16 + r + 8 * hi;
      out[(size_t)row * DIM + col] = c[r] + bv;
    }
  }
}

// ---------------------------------------------------------------------------
extern "C" void kernel_launch(void* const* d_in, const int* in_sizes, int n_in,
                              void* d_out, int out_size, void* d_ws,
                              size_t ws_size, hipStream_t stream) {
  (void)in_sizes; (void)n_in; (void)out_size; (void)ws_size;
  const float* x     = (const float*)d_in[0];
  const float* Wqkv  = (const float*)d_in[1];
  const float* Wproj = (const float*)d_in[2];
  const float* bproj = (const float*)d_in[3];
  float* out = (float*)d_out;

  _Float16* ws = (_Float16*)d_ws;
  const size_t perbuf = (size_t)BATCH * NHEAD * SEQ * HDIM;  // 6,291,456
  _Float16* q   = ws;
  _Float16* kT  = ws + perbuf;
  _Float16* v   = ws + 2 * perbuf;
  _Float16* ao  = ws + 3 * perbuf;            // MTOT*DIM == perbuf
  _Float16* x16 = ws + 4 * perbuf;            // MTOT*DIM
  _Float16* wq16 = x16 + perbuf;              // DIM*QKVN = 1,769,472
  _Float16* wp16 = wq16 + (size_t)DIM * QKVN; // DIM*DIM   =   589,824

  // 0) one-shot conversions (b128 in / b128 out, exact multiples of 2048)
  cvt_f16<<<(MTOT * DIM) / (256 * 8), 256, 0, stream>>>(x, x16);
  cvt_f16<<<(DIM * QKVN) / (256 * 8), 256, 0, stream>>>(Wqkv, wq16);
  cvt_f16<<<(DIM * DIM) / (256 * 8), 256, 0, stream>>>(Wproj, wp16);

  // 1) QKV: 36 x 128 blocks of 4 waves (16x64 tile each).
  qkv_gemm<<<dim3(QKVN / 64, MTOT / 16 / 4), 128, 0, stream>>>(x16, wq16, q, kT, v);
  // 2) Attention: 32 x 48 blocks of 4 waves.
  attention<<<dim3(SEQ / 16 / 4, BATCH * NHEAD), 128, 0, stream>>>(q, kT, v, ao);
  // 3) Projection: 12 x 128 blocks of 4 waves.
  proj_gemm<<<dim3(DIM / 64, MTOT / 16 / 4), 128, 0, stream>>>(ao, wp16, bproj, out);
}
